// LFA_84043920048548
// MI455X (gfx1250) — matrix operand order repaired and checked by reading, not compiled, not run on hardware
//
#include <hip/hip_runtime.h>
#include <hip/hip_bf16.h>

typedef __attribute__((ext_vector_type(16))) _Float16 v16h;
typedef __attribute__((ext_vector_type(8)))  _Float16 v8h;
typedef __attribute__((ext_vector_type(8)))  float    v8f;

#define LFA_B 2
#define LFA_N 50000
#define LFA_K 16
#define LFA_D 64
#define TOTAL_PTS (LFA_B * LFA_N)
#define WAVES_PER_BLOCK 8
#define LN_EPS 1e-5f

// half-element offsets of swizzled weight fragments inside the blob
#define OFF_G2 0        // 64x128  (2 ksteps x 8 ntiles)
#define OFF_G3 8192     // 128x64  (4 x 4)
#define OFF_F1 16384    // 128x64  (4 x 4)
#define OFF_F2 24576    // 64x128  (2 x 8)
#define OFF_F3 32768    // 128x64  (4 x 4)
#define WBLOB_HALFS 40960
#define PARAM_BYTE_OFF (WBLOB_HALFS * 2)            // 81920
#define PARAM_FLOATS 1536
#define BLOB_BYTES (PARAM_BYTE_OFF + PARAM_FLOATS * 4)  // 88064
#define BLOB_UINT4 (BLOB_BYTES / 16)                    // 5504
#define ACT_HALFS_PER_WAVE 2048                          // 16 x 128
#define SMEM_BYTES (BLOB_BYTES + WAVES_PER_BLOCK * ACT_HALFS_PER_WAVE * 2)

// float offsets within the param region
#define P_GW1 0
#define P_GB1 256
#define P_GLS1 320
#define P_GLB1 384
#define P_GB2 448
#define P_GLS2 576
#define P_GLB2 704
#define P_GB3 832
#define P_FB1 896
#define P_FLS1 960
#define P_FLB1 1024
#define P_FB2 1088
#define P_FLS2 1216
#define P_FLB2 1344
#define P_FB3 1472

// B-matrix (32x16 f16) fragment address for element (k, n) of a KxN weight.
// Contiguous k order used here (same as load_a): lanes 0-15 hold K=0..15, lanes 16-31 hold K=16..31,
// 2 halfs per VGPR in increasing K. Each lane's 16 halfs stored contiguously.
__device__ __forceinline__ int b_frag_addr(int k, int n, int ntiles) {
  int kstep = k >> 5, koff = k & 31;
  int nt = n >> 4, l16 = n & 15;
  int lane = ((koff >> 4) & 1) * 16 + l16;
  int hidx = koff & 15;
  return (kstep * ntiles + nt) * 512 + lane * 16 + hidx;
}

// ---------------------------------------------------------------- prep kernel
__global__ __launch_bounds__(256) void lfa_prep(
    void* ws,
    const float* gw1, const float* gb1, const float* gls1, const float* glb1,
    const float* gw2, const float* gb2, const float* gls2, const float* glb2,
    const float* gw3, const float* gb3,
    const float* fw1, const float* fb1, const float* fls1, const float* flb1,
    const float* fw2, const float* fb2, const float* fls2, const float* flb2,
    const float* fw3, const float* fb3) {
  _Float16* wh = (_Float16*)ws;
  float* pf = (float*)((char*)ws + PARAM_BYTE_OFF);
  int tid = blockIdx.x * blockDim.x + threadIdx.x;
  if (tid < 8192) {                       // g_w2: 64x128
    int k = tid >> 7, n = tid & 127;
    wh[OFF_G2 + b_frag_addr(k, n, 8)] = (_Float16)gw2[k * 128 + n];
  } else if (tid < 16384) {               // g_w3: 128x64
    int t = tid - 8192; int k = t >> 6, n = t & 63;
    wh[OFF_G3 + b_frag_addr(k, n, 4)] = (_Float16)gw3[k * 64 + n];
  } else if (tid < 24576) {               // f_w1: 128x64
    int t = tid - 16384; int k = t >> 6, n = t & 63;
    wh[OFF_F1 + b_frag_addr(k, n, 4)] = (_Float16)fw1[k * 64 + n];
  } else if (tid < 32768) {               // f_w2: 64x128
    int t = tid - 24576; int k = t >> 7, n = t & 127;
    wh[OFF_F2 + b_frag_addr(k, n, 8)] = (_Float16)fw2[k * 128 + n];
  } else if (tid < 40960) {               // f_w3: 128x64
    int t = tid - 32768; int k = t >> 6, n = t & 63;
    wh[OFF_F3 + b_frag_addr(k, n, 4)] = (_Float16)fw3[k * 64 + n];
  } else if (tid < 40960 + PARAM_FLOATS) {
    int t = tid - 40960;
    float v;
    if      (t < 256)  v = gw1[t];
    else if (t < 320)  v = gb1[t - 256];
    else if (t < 384)  v = gls1[t - 320];
    else if (t < 448)  v = glb1[t - 384];
    else if (t < 576)  v = gb2[t - 448];
    else if (t < 704)  v = gls2[t - 576];
    else if (t < 832)  v = glb2[t - 704];
    else if (t < 896)  v = gb3[t - 832];
    else if (t < 960)  v = fb1[t - 896];
    else if (t < 1024) v = fls1[t - 960];
    else if (t < 1088) v = flb1[t - 1024];
    else if (t < 1216) v = fb2[t - 1088];
    else if (t < 1344) v = fls2[t - 1216];
    else if (t < 1472) v = flb2[t - 1344];
    else               v = fb3[t - 1472];
    pf[t] = v;
  }
}

// ---------------------------------------------------------------- main kernel

// Lane XOR exchange via DS_SWIZZLE_B32, group-of-32 mode:
// offset = {1'b0, xor_mask[4:0], or_mask[4:0]=0, and_mask[4:0]=0x1f}
template <int MASK>
__device__ __forceinline__ float swz_xor(float v) {
  int i = __builtin_amdgcn_ds_swizzle(__float_as_int(v), (MASK << 10) | 0x1f);
  return __int_as_float(i);
}

// Butterfly sum across the 16-lane half-group (masks 1,2,4,8 stay in-group).
__device__ __forceinline__ float rsum16(float v) {
  v += swz_xor<1>(v);
  v += swz_xor<2>(v);
  v += swz_xor<4>(v);
  v += swz_xor<8>(v);
  return v;
}

// A-matrix 16x32 f16 fragment from row-major act tile (stride in halfs).
// Lane (hi,l16) is row m=l16; its 16 halfs are koff {hi*16..hi*16+15}:
// two contiguous 16B chunks.
__device__ __forceinline__ v16h load_a(const _Float16* act, int stride, int m,
                                       int s, int hi) {
  const v8h lo = *(const v8h*)(act + m * stride + s * 32 + hi * 16);
  const v8h hh = *(const v8h*)(act + m * stride + s * 32 + hi * 16 + 8);
  v16h a;
#pragma unroll
  for (int e = 0; e < 8; ++e) { a[e] = lo[e]; a[8 + e] = hh[e]; }
  return a;
}

__device__ __forceinline__ v16h load_b(const _Float16* frag, int lane) {
  const v8h* p = (const v8h*)(frag + lane * 16);
  v8h b0 = p[0], b1 = p[1];
  v16h b;
#pragma unroll
  for (int e = 0; e < 8; ++e) { b[e] = b0[e]; b[8 + e] = b1[e]; }
  return b;
}

template <int KS, int NT>
__device__ __forceinline__ void gemm(const _Float16* act, const _Float16* wfrag,
                                     const float* bias, int lane, int l16,
                                     int hi, v8f acc[NT]) {
#pragma unroll
  for (int t = 0; t < NT; ++t) {
    float bv = bias[t * 16 + l16];
#pragma unroll
    for (int r = 0; r < 8; ++r) acc[t][r] = bv;
  }
#pragma unroll
  for (int s = 0; s < KS; ++s) {
    v16h A = load_a(act, 128, l16, s, hi);
#pragma unroll
    for (int t = 0; t < NT; ++t) {
      v16h Bf = load_b(wfrag + (s * NT + t) * 512, lane);
      acc[t] = __builtin_amdgcn_wmma_f32_16x16x32_f16(
          false, A, false, Bf, (short)0, acc[t], false, false);
    }
  }
}

// LayerNorm over NT*16 channels + LeakyReLU, write f16 into act tile.
template <int NT>
__device__ __forceinline__ void ln_lrelu_store(v8f acc[NT], const float* gamma,
                                               const float* beta, _Float16* act,
                                               int l16, int hi) {
  const float invC = 1.0f / (float)(NT * 16);
  float mu[8], rs[8];
#pragma unroll
  for (int r = 0; r < 8; ++r) {
    float sv = 0.f, qv = 0.f;
#pragma unroll
    for (int t = 0; t < NT; ++t) { float x = acc[t][r]; sv += x; qv += x * x; }
    sv = rsum16(sv);
    qv = rsum16(qv);
    float m = sv * invC;
    mu[r] = m;
    rs[r] = rsqrtf(qv * invC - m * m + LN_EPS);
  }
#pragma unroll
  for (int t = 0; t < NT; ++t) {
    float g = gamma[t * 16 + l16], bb = beta[t * 16 + l16];
#pragma unroll
    for (int r = 0; r < 8; ++r) {
      float y = (acc[t][r] - mu[r]) * rs[r] * g + bb;
      y = y >= 0.f ? y : 0.2f * y;
      act[(hi * 8 + r) * 128 + t * 16 + l16] = (_Float16)y;
    }
  }
}

__global__ __launch_bounds__(256) void lfa_main(const float* __restrict__ pfeat,
                                                const float* __restrict__ geom,
                                                const int* __restrict__ nbr,
                                                const void* __restrict__ ws,
                                                float* __restrict__ out) {
  extern __shared__ __align__(16) char smem[];
  _Float16* wlds = (_Float16*)smem;
  float* plds = (float*)(smem + PARAM_BYTE_OFF);
  const int tid = threadIdx.x;

  {  // stage pre-swizzled weight/param blob into LDS
    const uint4* src = (const uint4*)ws;
    uint4* dst = (uint4*)smem;
    for (int i = tid; i < BLOB_UINT4; i += 256) dst[i] = src[i];
  }
  __syncthreads();

  const int warp = tid >> 5, lane = tid & 31, l16 = lane & 15, hi = lane >> 4;
  _Float16* act = (_Float16*)(smem + BLOB_BYTES) + warp * ACT_HALFS_PER_WAVE;
  const int pt = blockIdx.x * WAVES_PER_BLOCK + warp;  // one wave = one point
  if (pt >= TOTAL_PTS) return;                         // wave-uniform

  const int j = l16;  // neighbor row handled by this lane (both halves)

  // ---- g1: 4 -> 64 in VALU (f32 weights), LN(64)+LeakyReLU, f16 -> act[:, :64]
  {
    const float* xr = geom + ((size_t)pt * 16 + j) * 4;
    float x0 = xr[0], x1 = xr[1], x2 = xr[2], x3 = xr[3];
    const float* w1 = plds + P_GW1;  // [4][64]
    float h[32];
    float s = 0.f, q = 0.f;
#pragma unroll
    for (int c = 0; c < 32; ++c) {
      int col = hi * 32 + c;
      float v = w1[col] * x0 + w1[64 + col] * x1 + w1[128 + col] * x2 +
                w1[192 + col] * x3 + plds[P_GB1 + col];
      h[c] = v;
      s += v;
      q += v * v;
    }
    s += swz_xor<16>(s);
    q += swz_xor<16>(q);
    float mean = s * (1.0f / 64.0f);
    float rstd = rsqrtf(q * (1.0f / 64.0f) - mean * mean + LN_EPS);
#pragma unroll
    for (int u = 0; u < 4; ++u) {
      v8h hv;
#pragma unroll
      for (int e = 0; e < 8; ++e) {
        int col = hi * 32 + u * 8 + e;
        float y = (h[u * 8 + e] - mean) * rstd * plds[P_GLS1 + col] +
                  plds[P_GLB1 + col];
        y = y >= 0.f ? y : 0.2f * y;
        hv[e] = (_Float16)y;
      }
      *(v8h*)(act + j * 128 + hi * 32 + u * 8) = hv;
    }
  }

  // ---- g2: 64 -> 128 (WMMA), LN(128)+LeakyReLU
  {
    v8f acc[8];
    gemm<2, 8>(act, wlds + OFF_G2, plds + P_GB2, lane, l16, hi, acc);
    ln_lrelu_store<8>(acc, plds + P_GLS2, plds + P_GLB2, act, l16, hi);
  }

  // ---- g3: 128 -> 64 (WMMA), plain bias -> act[:, :64] as f16
  {
    v8f acc[4];
    gemm<4, 4>(act, wlds + OFF_G3, plds + P_GB3, lane, l16, hi, acc);
#pragma unroll
    for (int t = 0; t < 4; ++t)
#pragma unroll
      for (int r = 0; r < 8; ++r)
        act[(hi * 8 + r) * 128 + t * 16 + l16] = (_Float16)acc[t][r];
  }

  // ---- gather neighbor features -> act[:, 64:128] as f16 (L2-resident table)
  {
    const int nbi = nbr[(size_t)pt * 16 + j];
    const int bidx = pt / LFA_N;
    const float* src = pfeat + ((size_t)bidx * LFA_N + nbi) * 64 + hi * 32;
#pragma unroll
    for (int u = 0; u < 4; ++u) {
      float4 a = *(const float4*)(src + u * 8);
      float4 b = *(const float4*)(src + u * 8 + 4);
      v8h hv;
      hv[0] = (_Float16)a.x; hv[1] = (_Float16)a.y;
      hv[2] = (_Float16)a.z; hv[3] = (_Float16)a.w;
      hv[4] = (_Float16)b.x; hv[5] = (_Float16)b.y;
      hv[6] = (_Float16)b.z; hv[7] = (_Float16)b.w;
      *(v8h*)(act + j * 128 + 64 + hi * 32 + u * 8) = hv;
    }
  }

  // ---- f1: 128 -> 64 (WMMA), LN(64)+LeakyReLU -> act[:, :64]
  {
    v8f acc[4];
    gemm<4, 4>(act, wlds + OFF_F1, plds + P_FB1, lane, l16, hi, acc);
    ln_lrelu_store<4>(acc, plds + P_FLS1, plds + P_FLB1, act, l16, hi);
  }

  // ---- f2: 64 -> 128 (WMMA), LN(128)+LeakyReLU
  {
    v8f acc[8];
    gemm<2, 8>(act, wlds + OFF_F2, plds + P_FB2, lane, l16, hi, acc);
    ln_lrelu_store<8>(acc, plds + P_FLS2, plds + P_FLB2, act, l16, hi);
  }

  // ---- f3: 128 -> 64 (WMMA), then mean over the 16 rows (= K) -> out
  {
    v8f acc[4];
    gemm<4, 4>(act, wlds + OFF_F3, plds + P_FB3, lane, l16, hi, acc);
#pragma unroll
    for (int t = 0; t < 4; ++t) {
      float p = 0.f;
#pragma unroll
      for (int r = 0; r < 8; ++r) p += acc[t][r];
      p += swz_xor<16>(p);  // rows 0-7 + rows 8-15
      p *= (1.0f / 16.0f);
      if ((t >> 1) == hi) out[(size_t)pt * 64 + t * 16 + l16] = p;
    }
  }
}

// ---------------------------------------------------------------- launcher
extern "C" void kernel_launch(void* const* d_in, const int* in_sizes, int n_in,
                              void* d_out, int out_size, void* d_ws,
                              size_t ws_size, hipStream_t stream) {
  (void)in_sizes; (void)n_in; (void)out_size; (void)ws_size;
  const float* pfeat = (const float*)d_in[0];
  const float* geom  = (const float*)d_in[1];
  const int*   nbr   = (const int*)d_in[2];

  // prep: swizzle f16 weight fragments + pack params into d_ws
  const int prep_threads = 40960 + PARAM_FLOATS;  // 42496
  lfa_prep<<<(prep_threads + 255) / 256, 256, 0, stream>>>(
      d_ws,
      (const float*)d_in[3],  (const float*)d_in[4],  (const float*)d_in[5],
      (const float*)d_in[6],  (const float*)d_in[7],  (const float*)d_in[8],
      (const float*)d_in[9],  (const float*)d_in[10], (const float*)d_in[11],
      (const float*)d_in[12], (const float*)d_in[13], (const float*)d_in[14],
      (const float*)d_in[15], (const float*)d_in[16], (const float*)d_in[17],
      (const float*)d_in[18], (const float*)d_in[19], (const float*)d_in[20],
      (const float*)d_in[21], (const float*)d_in[22]);

  const int blocks = (TOTAL_PTS + WAVES_PER_BLOCK - 1) / WAVES_PER_BLOCK;
  lfa_main<<<blocks, 256, SMEM_BYTES, stream>>>(pfeat, geom, nbr, d_ws,
                                                (float*)d_out);
}
